// PolynormerAttention_32950989095698
// MI455X (gfx1250) — matrix operand order repaired and checked by reading, not compiled, not run on hardware
//
#include <hip/hip_runtime.h>

#define NROWS 100000
#define BETA 0.9f
#define EPS 1e-5f
#define NWG1 256
#define RB1 3125   // 32-row blocks in pass1
#define RB2 6250   // 16-row blocks in pass2

typedef __attribute__((ext_vector_type(4)))  unsigned int u32x4;
typedef __attribute__((ext_vector_type(4)))  float        f32x4;
typedef __attribute__((ext_vector_type(8)))  float        v8f;
typedef __attribute__((ext_vector_type(16))) __bf16       v16bf;

union Frag { u32x4 q[2]; v16bf v; };

__device__ __forceinline__ unsigned short f2bf(float f) {
  unsigned int u = __float_as_uint(f);
  u += 0x7FFFu + ((u >> 16) & 1u);           // round-to-nearest-even
  return (unsigned short)(u >> 16);
}
__device__ __forceinline__ float bf2f(unsigned short b) {
  return __uint_as_float(((unsigned int)b) << 16);
}
__device__ __forceinline__ float fast_rcp(float x) {
  return __builtin_amdgcn_rcpf(x);           // v_rcp_f32, ~1 ulp: plenty for bf16 data
}
__device__ __forceinline__ float fast_sigmoid(float x) {
  return fast_rcp(1.f + __expf(-x));
}
__device__ __forceinline__ v8f wmma_bf16(v16bf a, v16bf b, v8f c) {
  return __builtin_amdgcn_wmma_f32_16x16x32_bf16(false, a, false, b, (short)0, c,
                                                 false, false);
}
__device__ __forceinline__ v8f v8f_zero() {
  v8f z = {0.f, 0.f, 0.f, 0.f, 0.f, 0.f, 0.f, 0.f};
  return z;
}
// A fragment (16x32 bf16): row M = rowBase+(lane&15); K = kBase + {0..15} + 16*(lane>=16)
__device__ __forceinline__ v16bf loadA(const unsigned short* p0, int stride, int rowBase,
                                       int kBase, int lane) {
  const unsigned short* p = p0 + (rowBase + (lane & 15)) * stride + kBase + ((lane >> 4) << 4);
  Frag f;
  f.q[0] = *(const u32x4*)(p);
  f.q[1] = *(const u32x4*)(p + 8);
  return f.v;
}
// B fragment from packed layout: 16 contiguous bf16 per lane
__device__ __forceinline__ v16bf loadBpk(const unsigned short* tileBase, int lane) {
  const unsigned short* p = tileBase + lane * 16;
  Frag f;
  f.q[0] = *(const u32x4*)(p);
  f.q[1] = *(const u32x4*)(p + 8);
  return f.v;
}

// ---- weight repack: fp32 [512x512] (K-major rows) -> bf16 WMMA-B tiles ----
// tile id = tj*16+tc ; element lane*16+e = W[tc*32 + e + 16*(lane>=16)][tj*16 + (lane&15)]
__global__ void k_prep_weights(const float* __restrict__ hW, const float* __restrict__ kW,
                               const float* __restrict__ vW, const float* __restrict__ oW,
                               unsigned short* __restrict__ pk) {
  int t = blockIdx.x * 256 + threadIdx.x;      // 0 .. 4*2^18-1
  int mat = t >> 18;
  int rem = t & 262143;
  int tile = rem >> 9;
  int tc = tile & 15, tj = tile >> 4;
  int within = rem & 511;
  int lane = within >> 4, e = within & 15;
  int kk = tc * 32 + e + ((lane >> 4) << 4);
  int jj = tj * 16 + (lane & 15);
  const float* W = (mat == 0) ? hW : (mat == 1) ? kW : (mat == 2) ? vW : oW;
  pk[t] = f2bf(W[kk * 512 + jj]);
}

// ---- pass1: k = sigmoid(x@k_w), v = x@v_w, accumulate kv and k_sum partials ----
__global__ void __launch_bounds__(256) k_pass1(const float* __restrict__ x,
                                               const unsigned short* __restrict__ wpk_k,
                                               const unsigned short* __restrict__ wpk_v,
                                               float* __restrict__ kv_part,
                                               float* __restrict__ ksum_part) {
  extern __shared__ unsigned char smem[];
  unsigned short* xs = (unsigned short*)smem;   // [32][528] bf16 x block
  unsigned short* kT = xs + 32 * 528;           // [512][40] bf16, ch2-major (head de-interleaved)
  unsigned short* vT = kT + 512 * 40;           // [512][40]
  float* ksum = (float*)(vT + 512 * 40);        // [512] fp32

  const int tid = threadIdx.x;
  const int w = tid >> 5;          // wave id == head id for kv phase
  const int lane = tid & 31;
  const int hi8 = (lane >> 4) << 3;

  if (tid < 512) ksum[tid] = 0.f;
  v8f kvacc[4][4];
#pragma unroll
  for (int i = 0; i < 4; ++i)
#pragma unroll
    for (int j = 0; j < 4; ++j) kvacc[i][j] = v8f_zero();
  __syncthreads();

  for (int rb = blockIdx.x; rb < RB1; rb += gridDim.x) {
    const int n0 = rb * 32;
    // stage x block (coalesced float4 loads, convert to bf16)
    for (int i = tid; i < 4096; i += 256) {
      int r = i >> 7, c4 = (i & 127) << 2;
      f32x4 f = *(const f32x4*)(x + (size_t)(n0 + r) * 512 + c4);
      unsigned short* d = xs + r * 528 + c4;
      d[0] = f2bf(f[0]); d[1] = f2bf(f[1]); d[2] = f2bf(f[2]); d[3] = f2bf(f[3]);
    }
    // prefetch next block
    int rbn = rb + gridDim.x;
    if (rbn < RB1) {
      const float* pf = x + (size_t)rbn * 32 * 512 + tid * 64;
      __builtin_prefetch(pf, 0, 1);
      __builtin_prefetch(pf + 32, 0, 1);
    }
    __syncthreads();

    // k/v GEMMs: wave w covers output channels [w*64, w*64+64)
#pragma unroll
    for (int half = 0; half < 2; ++half) {
      v8f aK[4], aV[4];
#pragma unroll
      for (int j = 0; j < 4; ++j) { aK[j] = v8f_zero(); aV[j] = v8f_zero(); }
      for (int tc = 0; tc < 16; ++tc) {
        v16bf A = loadA(xs, 528, half * 16, tc * 32, lane);
#pragma unroll
        for (int j = 0; j < 4; ++j) {
          int jt = w * 4 + j;
          aK[j] = wmma_bf16(A, loadBpk(wpk_k + ((jt * 16 + tc) << 9), lane), aK[j]);
          aV[j] = wmma_bf16(A, loadBpk(wpk_v + ((jt * 16 + tc) << 9), lane), aV[j]);
        }
      }
#pragma unroll
      for (int j = 0; j < 4; ++j) {
        int c = (w * 4 + j) * 16 + (lane & 15);
        int ch2 = (c & 7) * 64 + (c >> 3);     // head-major channel: h*64+d
        float ssum = 0.f;
#pragma unroll
        for (int r = 0; r < 8; ++r) {
          int nn = half * 16 + r + hi8;
          float sg = fast_sigmoid(aK[j][r]);
          ssum += sg;
          kT[ch2 * 40 + nn] = f2bf(sg);
          vT[ch2 * 40 + nn] = f2bf(aV[j][r]);
        }
        atomicAdd(&ksum[ch2], ssum);           // ds_add_f32
      }
    }
    __syncthreads();

    // kv_h += k_h^T @ v_h (K = 32 rows), head h = w
#pragma unroll
    for (int i = 0; i < 4; ++i) {
      const unsigned short* pa = kT + (w * 64 + i * 16 + (lane & 15)) * 40 + (hi8 << 1);
      Frag fa; fa.q[0] = *(const u32x4*)pa; fa.q[1] = *(const u32x4*)(pa + 8);
#pragma unroll
      for (int j = 0; j < 4; ++j) {
        const unsigned short* pb = vT + (w * 64 + j * 16 + (lane & 15)) * 40 + (hi8 << 1);
        Frag fb; fb.q[0] = *(const u32x4*)pb; fb.q[1] = *(const u32x4*)(pb + 8);
        kvacc[i][j] = wmma_bf16(fa.v, fb.v, kvacc[i][j]);
      }
    }
    __syncthreads();
  }

  // spill per-WG partials
  float* kvp = kv_part + (size_t)blockIdx.x * 32768 + w * 4096;
#pragma unroll
  for (int i = 0; i < 4; ++i)
#pragma unroll
    for (int j = 0; j < 4; ++j)
#pragma unroll
      for (int r = 0; r < 8; ++r)
        kvp[(i * 16 + r + hi8) * 64 + j * 16 + (lane & 15)] = kvacc[i][j][r];
  __syncthreads();
  if (tid < 512) ksum_part[blockIdx.x * 512 + tid] = ksum[tid];
}

// ---- reduce partials -> kv (packed bf16 B-fragments) + k_sum (fp32) ----
__global__ void k_reduce(const float* __restrict__ kv_part, const float* __restrict__ ksum_part,
                         unsigned short* __restrict__ kv_pk, float* __restrict__ ksum_f) {
  int t = blockIdx.x * 256 + threadIdx.x;
  if (t < 32768) {
    int e = t & 15, lane = (t >> 4) & 31, tile = t >> 9;
    int jj = tile & 3, s = (tile >> 2) & 1, h = tile >> 3;
    int d = s * 32 + e + ((lane >> 4) << 4);
    int m = jj * 16 + (lane & 15);
    int src = h * 4096 + d * 64 + m;
    float acc = 0.f;
    for (int g = 0; g < NWG1; ++g) acc += kv_part[(size_t)g * 32768 + src];
    kv_pk[t] = f2bf(acc);
  } else if (t < 33280) {
    int j = t - 32768;
    float acc = 0.f;
    for (int g = 0; g < NWG1; ++g) acc += ksum_part[g * 512 + j];
    ksum_f[j] = acc;
  }
}

// ---- pass2: h & k GEMMs, num/den, layernorm*gate, out GEMM + relu ----
__global__ void __launch_bounds__(256) k_pass2(const float* __restrict__ x,
                                               const unsigned short* __restrict__ wpk_h,
                                               const unsigned short* __restrict__ wpk_k,
                                               const unsigned short* __restrict__ wpk_o,
                                               const unsigned short* __restrict__ kv_pk,
                                               const float* __restrict__ ksum_f,
                                               const float* __restrict__ h_b,
                                               const float* __restrict__ ln_g,
                                               const float* __restrict__ ln_b,
                                               const float* __restrict__ o_b,
                                               float* __restrict__ out) {
  extern __shared__ unsigned char smem[];
  unsigned short* xs = (unsigned short*)smem;   // [16][528] bf16 x block
  unsigned short* ks = xs + 16 * 528;           // [16][528] bf16 k (head-major channels)
  unsigned short* ys = ks + 16 * 528;           // [16][528] bf16 gated LN output
  float* hs  = (float*)(ys + 16 * 528);         // [16][520] fp32 h
  float* at  = hs + 16 * 520;                   // [16][520] fp32 attn
  float* den = at + 16 * 520;                   // [8][16]

  const int tid = threadIdx.x;
  const int w = tid >> 5, lane = tid & 31;
  const int hi8 = (lane >> 4) << 3;
  const int n0 = blockIdx.x * 16;

  for (int i = tid; i < 2048; i += 256) {
    int r = i >> 7, c4 = (i & 127) << 2;
    f32x4 f = *(const f32x4*)(x + (size_t)(n0 + r) * 512 + c4);
    unsigned short* d = xs + r * 528 + c4;
    d[0] = f2bf(f[0]); d[1] = f2bf(f[1]); d[2] = f2bf(f[2]); d[3] = f2bf(f[3]);
  }
  __syncthreads();

  // fused h and k GEMMs (shared A fragments)
  {
    v8f aH[4], aK[4];
#pragma unroll
    for (int j = 0; j < 4; ++j) { aH[j] = v8f_zero(); aK[j] = v8f_zero(); }
    for (int tc = 0; tc < 16; ++tc) {
      v16bf A = loadA(xs, 528, 0, tc * 32, lane);
#pragma unroll
      for (int j = 0; j < 4; ++j) {
        int jt = w * 4 + j;
        aH[j] = wmma_bf16(A, loadBpk(wpk_h + ((jt * 16 + tc) << 9), lane), aH[j]);
        aK[j] = wmma_bf16(A, loadBpk(wpk_k + ((jt * 16 + tc) << 9), lane), aK[j]);
      }
    }
#pragma unroll
    for (int j = 0; j < 4; ++j) {
      int c = (w * 4 + j) * 16 + (lane & 15);
      int ch2 = (c & 7) * 64 + (c >> 3);
      float hb = h_b[c];
#pragma unroll
      for (int r = 0; r < 8; ++r) {
        int row = r + hi8;
        hs[row * 520 + c] = aH[j][r] + hb;
        ks[row * 528 + ch2] = f2bf(fast_sigmoid(aK[j][r]));
      }
    }
  }
  __syncthreads();

  // per-head attention: den dot + num = q @ kv (WMMA, K=64 in two steps)
  {
    if (lane < 16) {
      float s = 0.f;
      for (int d = 0; d < 64; ++d)
        s += bf2f(ks[lane * 528 + w * 64 + d]) * ksum_f[w * 64 + d];
      den[w * 16 + lane] = s;
    }
    // per-lane reciprocal of den for the 8 rows this half-wave touches
    float rden[8];
#pragma unroll
    for (int r = 0; r < 8; ++r) rden[r] = fast_rcp(den[w * 16 + r + hi8]);
#pragma unroll
    for (int j = 0; j < 4; ++j) {
      v8f acc = v8f_zero();
#pragma unroll
      for (int s = 0; s < 2; ++s) {
        const unsigned short* pa = ks + (lane & 15) * 528 + w * 64 + s * 32 + (hi8 << 1);
        Frag fa; fa.q[0] = *(const u32x4*)pa; fa.q[1] = *(const u32x4*)(pa + 8);
        acc = wmma_bf16(fa.v, loadBpk(kv_pk + (((w * 2 + s) * 4 + j) << 9), lane), acc);
      }
      int m = j * 16 + (lane & 15);
#pragma unroll
      for (int r = 0; r < 8; ++r) {
        int row = r + hi8;
        at[row * 520 + m * 8 + w] = acc[r] * rden[r];   // channel c = m*8 + h
      }
    }
  }
  __syncthreads();

  // layernorm * (h + beta); each wave handles 2 rows, wave32 shfl reduction
#pragma unroll
  for (int rr = 0; rr < 2; ++rr) {
    int row = w * 2 + rr;
    float s = 0.f, s2 = 0.f;
#pragma unroll
    for (int i = 0; i < 16; ++i) {
      float v = at[row * 520 + lane + 32 * i];
      s += v; s2 += v * v;
    }
#pragma unroll
    for (int off = 16; off > 0; off >>= 1) {
      s += __shfl_xor(s, off, 32);
      s2 += __shfl_xor(s2, off, 32);
    }
    float mu = s * (1.f / 512.f);
    float var = s2 * (1.f / 512.f) - mu * mu;
    float rs = rsqrtf(var + EPS);
#pragma unroll
    for (int i = 0; i < 16; ++i) {
      int c = lane + 32 * i;
      float v = at[row * 520 + c];
      float y = ((v - mu) * rs * ln_g[c] + ln_b[c]) * (hs[row * 520 + c] + BETA);
      ys[row * 528 + c] = f2bf(y);
    }
  }
  __syncthreads();

  // out GEMM + bias + relu
  {
    v8f a[4];
#pragma unroll
    for (int j = 0; j < 4; ++j) a[j] = v8f_zero();
    for (int tc = 0; tc < 16; ++tc) {
      v16bf A = loadA(ys, 528, 0, tc * 32, lane);
#pragma unroll
      for (int j = 0; j < 4; ++j)
        a[j] = wmma_bf16(A, loadBpk(wpk_o + (((w * 4 + j) * 16 + tc) << 9), lane), a[j]);
    }
#pragma unroll
    for (int j = 0; j < 4; ++j) {
      int c = (w * 4 + j) * 16 + (lane & 15);
      float ob = o_b[c];
#pragma unroll
      for (int r = 0; r < 8; ++r) {
        int row = r + hi8;
        float v = a[j][r] + ob;
        out[(size_t)(n0 + row) * 512 + c] = v > 0.f ? v : 0.f;
      }
    }
  }
}

extern "C" void kernel_launch(void* const* d_in, const int* in_sizes, int n_in,
                              void* d_out, int out_size, void* d_ws, size_t ws_size,
                              hipStream_t stream) {
  const float* x    = (const float*)d_in[0];
  const float* h_w  = (const float*)d_in[1];
  const float* h_b  = (const float*)d_in[2];
  const float* k_w  = (const float*)d_in[3];
  const float* v_w  = (const float*)d_in[4];
  const float* ln_g = (const float*)d_in[5];
  const float* ln_b = (const float*)d_in[6];
  const float* o_w  = (const float*)d_in[7];
  const float* o_b  = (const float*)d_in[8];
  float* out = (float*)d_out;

  unsigned char* ws = (unsigned char*)d_ws;
  unsigned short* wpk    = (unsigned short*)ws;                          // 4 * 262144 bf16 (2 MB)
  float* kv_part         = (float*)(ws + (size_t)2097152);               // 256*32768 f32 (32 MB)
  float* ksum_part       = (float*)(ws + (size_t)2097152 + 33554432);    // 256*512 f32 (0.5 MB)
  unsigned short* kv_pk  = (unsigned short*)(ws + (size_t)2097152 + 33554432 + 524288); // 32768 bf16
  float* ksum_f          = (float*)(ws + (size_t)2097152 + 33554432 + 524288 + 65536);  // 512 f32

  const size_t smem1 = 32 * 528 * 2 + 2 * 512 * 40 * 2 + 512 * 4;        // 117760 B
  const size_t smem2 = 3 * 16 * 528 * 2 + 2 * 16 * 520 * 4 + 128 * 4;    // 117760 B

  k_prep_weights<<<4096, 256, 0, stream>>>(h_w, k_w, v_w, o_w, wpk);
  k_pass1<<<NWG1, 256, smem1, stream>>>(x, wpk + 262144, wpk + 2 * 262144, kv_part, ksum_part);
  k_reduce<<<131, 256, 0, stream>>>(kv_part, ksum_part, kv_pk, ksum_f);
  k_pass2<<<RB2, 256, smem2, stream>>>(x, wpk, wpk + 262144, wpk + 3 * 262144, kv_pk, ksum_f,
                                       h_b, ln_g, ln_b, o_b, out);
}
